// st_attention_block_59691455479828
// MI455X (gfx1250) — matrix operand order repaired and checked by reading, not compiled, not run on hardware
//
#include <hip/hip_runtime.h>

typedef _Float16 v16h __attribute__((ext_vector_type(16)));
typedef _Float16 v8h  __attribute__((ext_vector_type(8)));
typedef float    v8f  __attribute__((ext_vector_type(8)));

#define N_   64
#define CIN  64
#define COUT 64
#define R_   8
#define T_   256
#define V_   25

// ---------------------------------------------------------------------------
// Kernel 0: xm[n,c,v] = mean_t x[n,c,t,v].  One block per (n,c); wave = t-group,
// lanes = v (coalesced 100B rows).
// ---------------------------------------------------------------------------
__global__ __launch_bounds__(256) void k0_mean(const float* __restrict__ x,
                                               float* __restrict__ xm) {
  const int bx  = blockIdx.x;            // n*64 + c
  const int tid = threadIdx.x;
  const int tg  = tid >> 5;              // 8 t-groups (one per wave)
  const int v   = tid & 31;
  const float* p = x + (size_t)bx * (T_ * V_);
  __shared__ float red[8][V_];
  float s = 0.f;
  if (v < V_) {
    for (int t = tg; t < T_; t += 8) s += p[t * V_ + v];
    red[tg][v] = s;
  }
  __syncthreads();
  if (tid < V_) {
    float tot = 0.f;
    for (int g = 0; g < 8; ++g) tot += red[g][tid];
    xm[(size_t)bx * V_ + tid] = tot * (1.0f / T_);
  }
}

// ---------------------------------------------------------------------------
// Kernel 1: per n — k/q projections, tanh attention, fold Wf/bf/alpha/adjc into
// a[n,o,v,w]; store f16 padded to [n][o][32][32] (zeros in pad) for WMMA B-frags.
// ---------------------------------------------------------------------------
__global__ __launch_bounds__(256) void k1_attn(
    const float* __restrict__ xm, const float* __restrict__ adjc,
    const float* __restrict__ Wk, const float* __restrict__ bk,
    const float* __restrict__ Wq, const float* __restrict__ bq,
    const float* __restrict__ Wf, const float* __restrict__ bf,
    const float* __restrict__ alpha_dp, const float* __restrict__ beta_dp,
    const float* __restrict__ alpha, _Float16* __restrict__ a16) {
  const int n = blockIdx.x, tid = threadIdx.x;
  __shared__ float sXm[CIN * V_];
  __shared__ float sK[R_ * V_], sQ[R_ * V_];
  __shared__ float sAtt[R_ * V_ * V_];
  __shared__ float sWk[R_ * CIN], sWq[R_ * CIN], sWf[COUT * R_];
  __shared__ float sBk[R_], sBq[R_], sBf[COUT];

  for (int i = tid; i < CIN * V_; i += 256) sXm[i] = xm[(size_t)n * (CIN * V_) + i];
  for (int i = tid; i < R_ * CIN; i += 256) { sWk[i] = Wk[i]; sWq[i] = Wq[i]; }
  for (int i = tid; i < COUT * R_; i += 256) sWf[i] = Wf[i];
  if (tid < R_)   { sBk[tid] = bk[tid]; sBq[tid] = bq[tid]; }
  if (tid < COUT) sBf[tid] = bf[tid];
  __syncthreads();

  if (tid < R_ * V_) {
    const int r = tid / V_, v = tid % V_;
    float sk = 0.f, sq = 0.f;
    for (int c = 0; c < CIN; ++c) {
      const float xv = sXm[c * V_ + v];
      sk += sWk[r * CIN + c] * xv;
      sq += sWq[r * CIN + c] * xv;
    }
    sK[tid] = sk + sBk[r];
    sQ[tid] = sq + sBq[r];
  }
  __syncthreads();

  const float adp = alpha_dp[0], bdp = beta_dp[0], al = alpha[0];
  for (int i = tid; i < R_ * V_ * V_; i += 256) {
    const int r = i / (V_ * V_), rem = i % (V_ * V_);
    const int v = rem / V_, w = rem % V_;
    const float kk = sK[r * V_ + v], qq = sQ[r * V_ + w];
    sAtt[i] = tanhf(adp * kk * qq + bdp * (kk - qq));
  }
  __syncthreads();

  _Float16* outp = a16 + (size_t)n * (COUT * 32 * 32);
  for (int i = tid; i < COUT * 32 * 32; i += 256) {
    const int o = i >> 10, rem = i & 1023;
    const int v = rem >> 5, w = rem & 31;
    float val = 0.f;
    if (v < V_ && w < V_) {
      float s = sBf[o];
      for (int r = 0; r < R_; ++r) s += sWf[o * R_ + r] * sAtt[r * (V_ * V_) + v * V_ + w];
      val = al * s + adjc[v * V_ + w];
    }
    outp[i] = (_Float16)val;
  }
}

// ---------------------------------------------------------------------------
// Kernel 2: fused value-projection + attention-apply with f16 WMMA.
// Block = (t-tile of 16, n).  GEMM1: vmat[o,t,w] = Wv@x (+bv), GEMM2:
// out[o,t,v] = vmat[o,t,:] . a16[o,v,:].
// ---------------------------------------------------------------------------
#define XT_STRIDE 72   // halves; 144B row: 16B aligned chunks, conflict-free
#define WV_STRIDE 72
#define VM_STRIDE 40   // halves; 80B row: 16B aligned chunks

__global__ __launch_bounds__(256) void k2_main(
    const float* __restrict__ x, const float* __restrict__ Wv,
    const float* __restrict__ bv, const _Float16* __restrict__ a16,
    float* __restrict__ out) {
  const int t0   = blockIdx.x * 16;     // 16 t-tiles
  const int n    = blockIdx.y;          // 64
  const int tid  = threadIdx.x;
  // wave index is wave-uniform: pin it to an SGPR so the tile loop and all
  // per-wave o/address math compile as scalar (full-EXEC loop, SALU addressing).
  const int wave = __builtin_amdgcn_readfirstlane(tid >> 5);
  const int lane = tid & 31;
  const int h    = lane >> 4;           // lane half
  const int lp   = lane & 15;

  __shared__ __align__(16) _Float16 sXT[400 * XT_STRIDE];  // x^T tile [tw][c] f16
  __shared__ __align__(16) _Float16 sWv[64 * WV_STRIDE];   // Wv [o][c] f16
  __shared__ __align__(16) _Float16 sVm[16 * 16 * VM_STRIDE]; // vmat [o_rel][t][w]

  // ---- stage x tile (f32 -> f16, transpose to K-contiguous) ----
  const float* xb = x + (size_t)(n * CIN) * (T_ * V_) + t0 * V_;
  for (int i = tid; i < CIN * 400; i += 256) {
    const int c = i / 400, tv = i % 400;
    sXT[tv * XT_STRIDE + c] = (_Float16)xb[(size_t)c * (T_ * V_) + tv];
  }
  // ---- stage Wv (f32 -> f16) ----
  for (int i = tid; i < COUT * CIN; i += 256)
    sWv[(i >> 6) * WV_STRIDE + (i & 63)] = (_Float16)Wv[i];
  // ---- zero the w=25..39 pad of vmat once (never overwritten) ----
  for (int i = tid; i < 256 * (VM_STRIDE - V_); i += 256) {
    const int row = i / (VM_STRIDE - V_), w = V_ + i % (VM_STRIDE - V_);
    sVm[row * VM_STRIDE + w] = (_Float16)0.f;
  }

  for (int og = 0; og < 4; ++og) {
    const int ob = og * 16;
    __syncthreads();   // xT/Wv ready (og=0); prev GEMM2 reads of sVm done (og>0)

    // ================= GEMM1: vmat = Wv @ x  (M=o, N=tw, K=c) =================
    {
      // A fragments (Wv rows, contiguous K = 16h..16h+15, matching B)
      const int rowW = (ob + lp) * WV_STRIDE;
      const v8h a0l = *(const v8h*)&sWv[rowW + 16 * h];
      const v8h a0h = *(const v8h*)&sWv[rowW + 16 * h + 8];
      const v8h a1l = *(const v8h*)&sWv[rowW + 32 + 16 * h];
      const v8h a1h = *(const v8h*)&sWv[rowW + 32 + 16 * h + 8];
      const v16h A0 = __builtin_shufflevector(a0l, a0h, 0,1,2,3,4,5,6,7,8,9,10,11,12,13,14,15);
      const v16h A1 = __builtin_shufflevector(a1l, a1h, 0,1,2,3,4,5,6,7,8,9,10,11,12,13,14,15);
      float bvr[8];
      for (int r = 0; r < 8; ++r) bvr[r] = bv[ob + r + 8 * h];

      for (int tile = wave; tile < 25; tile += 8) {   // scalar (SGPR) loop
        const int tw   = tile * 16 + lp;
        const int rowX = tw * XT_STRIDE;
        // B fragments (x^T rows; contiguous K = 16h..16h+15, same order as A)
        const v8h b0l = *(const v8h*)&sXT[rowX + 16 * h];
        const v8h b0h = *(const v8h*)&sXT[rowX + 16 * h + 8];
        const v8h b1l = *(const v8h*)&sXT[rowX + 32 + 16 * h];
        const v8h b1h = *(const v8h*)&sXT[rowX + 32 + 16 * h + 8];
        const v16h B0 = __builtin_shufflevector(b0l, b0h, 0,1,2,3,4,5,6,7,8,9,10,11,12,13,14,15);
        const v16h B1 = __builtin_shufflevector(b1l, b1h, 0,1,2,3,4,5,6,7,8,9,10,11,12,13,14,15);
        v8f acc = {};
        acc = __builtin_amdgcn_wmma_f32_16x16x32_f16(false, A0, false, B0,
                                                     (short)0, acc, false, false);
        acc = __builtin_amdgcn_wmma_f32_16x16x32_f16(false, A1, false, B1,
                                                     (short)0, acc, false, false);
        const int tt = tw / 25, w = tw % 25;   // D col N=tw -> (t, w)
        for (int r = 0; r < 8; ++r) {          // D row M = r + 8h = o_rel
          sVm[((r + 8 * h) * 16 + tt) * VM_STRIDE + w] = (_Float16)(acc[r] + bvr[r]);
        }
      }
    }
    __syncthreads();   // vmat ready

    // ============ GEMM2: out[o,t,v] = sum_w vmat[o,t,w] * a16[o,v,w] ============
    for (int j = 0; j < 2; ++j) {
      const int orel = (wave << 1) | j;       // scalar: 16 o's / 8 waves
      const int o    = ob + orel;
      // A fragment: vmat rows (M=t=lp, K=w)
      const int rowV = (orel * 16 + lp) * VM_STRIDE;
      const v8h al_  = *(const v8h*)&sVm[rowV + 16 * h];
      const v8h ah_  = *(const v8h*)&sVm[rowV + 16 * h + 8];
      const v16h A   = __builtin_shufflevector(al_, ah_, 0,1,2,3,4,5,6,7,8,9,10,11,12,13,14,15);
      // B fragments from global a16 (row v, contiguous K = 16h..16h+15), zero-padded
      const _Float16* ga = a16 + (size_t)(n * COUT + o) * 1024;
      const _Float16* g0 = ga + (size_t)lp * 32 + 16 * h;          // v = lp
      const _Float16* g1 = ga + (size_t)(16 + lp) * 32 + 16 * h;   // v = 16+lp
      const v8h b0l = *(const v8h*)g0;
      const v8h b0h = *(const v8h*)(g0 + 8);
      const v8h b1l = *(const v8h*)g1;
      const v8h b1h = *(const v8h*)(g1 + 8);
      const v16h B0 = __builtin_shufflevector(b0l, b0h, 0,1,2,3,4,5,6,7,8,9,10,11,12,13,14,15);
      const v16h B1 = __builtin_shufflevector(b1l, b1h, 0,1,2,3,4,5,6,7,8,9,10,11,12,13,14,15);
      const v8f z = {};
      const v8f D0 = __builtin_amdgcn_wmma_f32_16x16x32_f16(false, A, false, B0,
                                                            (short)0, z, false, false);
      const v8f D1 = __builtin_amdgcn_wmma_f32_16x16x32_f16(false, A, false, B1,
                                                            (short)0, z, false, false);
      float* op = out + ((size_t)(n * COUT + o) * T_ + t0) * V_;
      for (int r = 0; r < 8; ++r) {
        const int t = r + 8 * h;              // D row M
        op[t * V_ + lp] = D0[r];              // v = lp (0..15)
        if (lp < 9) op[t * V_ + 16 + lp] = D1[r];   // v = 16..24
      }
    }
  }
}

// ---------------------------------------------------------------------------
extern "C" void kernel_launch(void* const* d_in, const int* in_sizes, int n_in,
                              void* d_out, int out_size, void* d_ws, size_t ws_size,
                              hipStream_t stream) {
  const float* x        = (const float*)d_in[0];
  const float* adjc     = (const float*)d_in[1];
  const float* Wk       = (const float*)d_in[2];
  const float* bk       = (const float*)d_in[3];
  const float* Wq       = (const float*)d_in[4];
  const float* bq       = (const float*)d_in[5];
  const float* Wv       = (const float*)d_in[6];
  const float* bv       = (const float*)d_in[7];
  const float* Wf       = (const float*)d_in[8];
  const float* bf       = (const float*)d_in[9];
  const float* alpha_dp = (const float*)d_in[10];
  const float* beta_dp  = (const float*)d_in[11];
  const float* alpha    = (const float*)d_in[12];
  float* out = (float*)d_out;

  // workspace layout: xm (N*C*V f32) | a16 (N*O*32*32 f16)
  float*    xm  = (float*)d_ws;
  _Float16* a16 = (_Float16*)((char*)d_ws + (size_t)N_ * CIN * V_ * sizeof(float));

  k0_mean<<<N_ * CIN, 256, 0, stream>>>(x, xm);
  k1_attn<<<N_, 256, 0, stream>>>(xm, adjc, Wk, bk, Wq, bq, Wf, bf,
                                  alpha_dp, beta_dp, alpha, a16);
  k2_main<<<dim3(T_ / 16, N_), 256, 0, stream>>>(x, Wv, bv, a16, out);
}
